// MLPEdgeDecoder_29008209117686
// MI455X (gfx1250) — matrix operand order repaired and checked by reading, not compiled, not run on hardware
//
#include <hip/hip_runtime.h>
#include <hip/hip_bf16.h>
#include <stdint.h>

typedef __attribute__((ext_vector_type(16))) __bf16 v16bf;
typedef __attribute__((ext_vector_type(2)))  __bf16 v2bf;
typedef __attribute__((ext_vector_type(8)))  float  v8f;

struct U8x32 { uint32_t x[8]; };   // 32 bytes, bit-cast target for v16bf

#define LATENT 64
#define HIDDEN 32
#define IN_DIM 128

// --- accurate RNE pack (prep kernel only; runs once over 8 KB) -------------
__device__ __forceinline__ uint32_t bf16_bits_rne(float f) {
    uint32_t u = __float_as_uint(f);
    return (u + 0x7FFFu + ((u >> 16) & 1u)) >> 16;
}
__device__ __forceinline__ uint32_t pack2_rne(float lo, float hi) {
    return bf16_bits_rne(lo) | (bf16_bits_rne(hi) << 16);
}

// --- fast pack for the hot gather->bf16 path -------------------------------
__device__ __forceinline__ uint32_t pack2_fast(float lo, float hi) {
#if __has_builtin(__builtin_amdgcn_cvt_pk_bf16_f32)
    v2bf p = __builtin_amdgcn_cvt_pk_bf16_f32(lo, hi);      // 1 VALU op
    return __builtin_bit_cast(uint32_t, p);
#elif __has_builtin(__builtin_amdgcn_perm)
    // round-half-up, then splice the two high halves in one v_perm_b32
    uint32_t ul = __float_as_uint(lo) + 0x8000u;
    uint32_t uh = __float_as_uint(hi) + 0x8000u;
    return __builtin_amdgcn_perm(uh, ul, 0x07060302u);      // 3 VALU ops
#else
    uint32_t ul = __float_as_uint(lo) + 0x8000u;
    uint32_t uh = __float_as_uint(hi) + 0x8000u;
    return (ul >> 16) | (uh & 0xFFFF0000u);
#endif
}

// --- fast tanh via native transcendentals ----------------------------------
__device__ __forceinline__ float fast_tanh(float x) {
#if __has_builtin(__builtin_amdgcn_exp2f) && __has_builtin(__builtin_amdgcn_rcpf)
#if __has_builtin(__builtin_amdgcn_fmed3f)
    x = __builtin_amdgcn_fmed3f(x, -9.0f, 9.0f);            // keep exp2 finite
#else
    x = fminf(fmaxf(x, -9.0f), 9.0f);
#endif
    float y = __builtin_amdgcn_exp2f(x * 2.8853900817779268f); // 2*log2(e)
    return (y - 1.0f) * __builtin_amdgcn_rcpf(y + 1.0f);
#else
    return tanhf(x);
#endif
}

// ---------------------------------------------------------------------------
// Prep: W1 [128][32] f32 -> bf16 in WMMA B-matrix lane layout, packed u32.
// Slot s = ((t*4 + kb)*32 + lane)*8 + j holds W1[k..k+1][n] with
//   i = 2j, k = kb*32 + (lane>=16 ? 16 : 0) + i, n = (lane&15) + 16*t
// (contiguous k order, matched by A: VGPR j: lanes 0-15 K=2j,2j+1; lanes 16-31 K=16+2j,..)
// Total 2048 u32 = 8 KB in d_ws.
// ---------------------------------------------------------------------------
__global__ void prep_w1_kernel(const float* __restrict__ W1,
                               uint32_t* __restrict__ wsb) {
    int s = blockIdx.x * blockDim.x + threadIdx.x;
    if (s >= 2048) return;
    int j    = s & 7;
    int lane = (s >> 3) & 31;
    int kb   = (s >> 8) & 3;
    int t    = s >> 10;
    int i    = j * 2;
    int k    = kb * 32 + ((lane & 16) ? 16 : 0) + i;
    int n    = (lane & 15) + 16 * t;
    float lo = W1[k * HIDDEN + n];
    float hi = W1[(k + 1) * HIDDEN + n];
    wsb[s] = pack2_rne(lo, hi);
}

// ---------------------------------------------------------------------------
// Main: one wave32 per 16-edge tile, grid-strided so each wave amortizes the
// register-resident W1-bf16 B tiles (64 VGPRs) over many tiles.
//   h[16x32] = bf16WMMA( concat(z[src],z[dst])[16x128] , W1bf[128x32] )
//   out = tanh( leaky(h + b1) @ W2 + b2 )
// ---------------------------------------------------------------------------
__global__ __launch_bounds__(256) void edge_mlp_kernel(
    const float* __restrict__ z, const int* __restrict__ eidx,
    const float* __restrict__ b1, const float* __restrict__ W2,
    const float* __restrict__ b2, const uint32_t* __restrict__ wsb,
    float* __restrict__ out, int n_edges) {

    const int lane    = threadIdx.x & 31;
    const int wave    = (int)((blockIdx.x * blockDim.x + threadIdx.x) >> 5);
    const int nwaves  = (int)((gridDim.x * blockDim.x) >> 5);
    const int n_tiles = (n_edges + 15) >> 4;

    const int m      = lane & 15;          // row within tile (both half-waves)
    const int hiHalf = (lane >> 4) & 1;    // lanes 16-31

    // ---- load all B tiles once: 2 N-tiles x 4 K-steps = 64 VGPRs ----------
    const uint4* bw = (const uint4*)wsb;
    v16bf B[2][4];
#pragma unroll
    for (int t = 0; t < 2; ++t)
#pragma unroll
        for (int kb = 0; kb < 4; ++kb) {
            const uint4* bp = bw + (size_t)(((t * 4 + kb) * 32 + lane) * 2);
            uint4 q0 = bp[0], q1 = bp[1];
            U8x32 bu;
            bu.x[0]=q0.x; bu.x[1]=q0.y; bu.x[2]=q0.z; bu.x[3]=q0.w;
            bu.x[4]=q1.x; bu.x[5]=q1.y; bu.x[6]=q1.z; bu.x[7]=q1.w;
            B[t][kb] = __builtin_bit_cast(v16bf, bu);
        }

    // ---- per-lane epilogue constants (hoisted out of tile loop) -----------
    const float b1lo = b1[m];
    const float b1hi = b1[m + 16];
    const float w2lo = W2[m];
    const float w2hi = W2[m + 16];
    const float bb   = b2[0];

    for (int tile = wave; tile < n_tiles; tile += nwaves) {
        int e = tile * 16 + m;
        if (e >= n_edges) e = n_edges - 1;     // tail clamp (gather only)
        const int src = eidx[e];
        const int dst = eidx[n_edges + e];
        const float* zs = z + (size_t)src * LATENT;
        const float* zd = z + (size_t)dst * LATENT;

        v8f c0 = {};   // hidden columns 0..15
        v8f c1 = {};   // hidden columns 16..31

#pragma unroll
        for (int kb = 0; kb < 4; ++kb) {
            // A-layout (16-bit 16x32): lane covers two 8-wide K runs;
            // run source (zs vs zd) is compile-time per kb (runs are 8-aligned,
            // kb 0-1 -> src half, kb 2-3 -> dst half).
            const int kbase = kb * 32 + hiHalf * 16;
            float4 f[4];
#pragma unroll
            for (int r = 0; r < 2; ++r) {
                const int K0 = kbase + r * 8;
                const float* p = (K0 < LATENT) ? (zs + K0) : (zd + (K0 - LATENT));
                f[2 * r]     = *(const float4*)(p);
                f[2 * r + 1] = *(const float4*)(p + 4);
            }
            U8x32 au;
            au.x[0] = pack2_fast(f[0].x, f[0].y);  au.x[1] = pack2_fast(f[0].z, f[0].w);
            au.x[2] = pack2_fast(f[1].x, f[1].y);  au.x[3] = pack2_fast(f[1].z, f[1].w);
            au.x[4] = pack2_fast(f[2].x, f[2].y);  au.x[5] = pack2_fast(f[2].z, f[2].w);
            au.x[6] = pack2_fast(f[3].x, f[3].y);  au.x[7] = pack2_fast(f[3].z, f[3].w);
            v16bf a = __builtin_bit_cast(v16bf, au);

            c0 = __builtin_amdgcn_wmma_f32_16x16x32_bf16(false, a, false, B[0][kb],
                                                         (short)0, c0, false, false);
            c1 = __builtin_amdgcn_wmma_f32_16x16x32_bf16(false, a, false, B[1][kb],
                                                         (short)0, c1, false, false);
        }

        // Epilogue: bias + LeakyReLU + second linear (per-lane column), then
        // half-wave reduction over the 32 hidden columns (2 per lane) this half holds.
        // C layout: VGPR r -> row M=r (lanes 0-15) / M=8+r (lanes 16-31), N=lane&15.
        float pr[8];
#pragma unroll
        for (int r = 0; r < 8; ++r) {
            float h0 = c0[r] + b1lo; h0 = (h0 >= 0.f) ? h0 : 0.01f * h0;
            float h1 = c1[r] + b1hi; h1 = (h1 >= 0.f) ? h1 : 0.01f * h1;
            pr[r] = h0 * w2lo + h1 * w2hi;
        }
#pragma unroll
        for (int r = 0; r < 8; ++r) {
            float v = pr[r];
            v += __shfl_xor(v, 1);   // masks < 16 stay within each half-wave
            v += __shfl_xor(v, 2);
            v += __shfl_xor(v, 4);
            v += __shfl_xor(v, 8);
            pr[r] = v;
        }

        if (m == 0) {
            const int base = tile * 16 + hiHalf * 8;   // rows 0-7 or 8-15
            if (base + 8 <= n_edges) {
                float4 o0 = { fast_tanh(pr[0] + bb), fast_tanh(pr[1] + bb),
                              fast_tanh(pr[2] + bb), fast_tanh(pr[3] + bb) };
                float4 o1 = { fast_tanh(pr[4] + bb), fast_tanh(pr[5] + bb),
                              fast_tanh(pr[6] + bb), fast_tanh(pr[7] + bb) };
                *(float4*)(out + base)     = o0;
                *(float4*)(out + base + 4) = o1;
            } else {
#pragma unroll
                for (int r = 0; r < 8; ++r)
                    if (base + r < n_edges) out[base + r] = fast_tanh(pr[r] + bb);
            }
        }
    }
}

extern "C" void kernel_launch(void* const* d_in, const int* in_sizes, int n_in,
                              void* d_out, int out_size, void* d_ws, size_t ws_size,
                              hipStream_t stream) {
    const float* z   = (const float*)d_in[0];
    const int*   eix = (const int*)  d_in[1];
    const float* W1  = (const float*)d_in[2];
    const float* b1  = (const float*)d_in[3];
    const float* W2  = (const float*)d_in[4];
    const float* b2  = (const float*)d_in[5];
    float*       out = (float*)d_out;
    uint32_t*    wsb = (uint32_t*)d_ws;   // 8 KB used

    const int n_edges = in_sizes[1] / 2;

    // Pack W1 into bf16 WMMA B layout (tiny; same stream => ordered).
    hipLaunchKernelGGL(prep_w1_kernel, dim3(8), dim3(256), 0, stream, W1, wsb);

    // ~8 tiles per wave so register-resident B amortizes over the gathers.
    const int n_tiles = (n_edges + 15) / 16;
    const int waves_per_block = 256 / 32;
    const int tiles_per_wave  = 8;
    int blocks = (n_tiles + waves_per_block * tiles_per_wave - 1) /
                 (waves_per_block * tiles_per_wave);
    if (blocks < 1) blocks = 1;
    hipLaunchKernelGGL(edge_mlp_kernel, dim3(blocks), dim3(256), 0, stream,
                       z, eix, b1, W2, b2, wsb, out, n_edges);
}
